// GroupQueryAttn_12773232738992
// MI455X (gfx1250) — matrix operand order repaired and checked by reading, not compiled, not run on hardware
//
#include <hip/hip_runtime.h>
#include <hip/hip_bf16.h>
#include <stdint.h>

#define HIDDEN 2048
#define SEQ 2048
#define BATCH 2
#define NUM_HEAD 32
#define NUM_KV_HEAD 8
#define HEAD_DIM 64
#define KV_DIM 512
#define MTOT (BATCH * SEQ) // 4096

typedef __bf16 bf16_t;
typedef bf16_t v16bf __attribute__((ext_vector_type(16)));
typedef float v8f __attribute__((ext_vector_type(8)));

union AFrag {
    v16bf v;
    unsigned short u[16];
    unsigned int w[8];
};

__device__ __forceinline__ unsigned short f2bf(float f) {
    unsigned int u = __float_as_uint(f);
    u += 0x7FFFu + ((u >> 16) & 1u); // round-to-nearest-even
    return (unsigned short)(u >> 16);
}

// LDS byte offset of a shared-memory pointer (flat addr low 32 bits).
__device__ __forceinline__ unsigned lds_off(const void* p) {
    return (unsigned)(uintptr_t)p;
}

// CDNA5 async global->LDS 16B copy, tracked by ASYNCcnt.
__device__ __forceinline__ void async_copy_b128(unsigned lds_addr,
                                                unsigned long long gaddr) {
    asm volatile("global_load_async_to_lds_b128 %0, %1, off"
                 :
                 : "v"(lds_addr), "v"(gaddr)
                 : "memory");
}

__device__ __forceinline__ void wait_async0() {
    asm volatile("s_wait_asynccnt 0x0" ::: "memory");
}

// ---------------------------------------------------------------------------
// f32 -> bf16 elementwise convert
// ---------------------------------------------------------------------------
__global__ void cvt_f32_bf16(const float* __restrict__ in,
                             unsigned short* __restrict__ out, int n) {
    int i = blockIdx.x * blockDim.x + threadIdx.x;
    if (i < n) out[i] = f2bf(in[i]);
}

// ---------------------------------------------------------------------------
// W[k][n] (f32, K x N) -> Wt[n][k] (bf16, N x K)
// ---------------------------------------------------------------------------
__global__ void transpose_cvt(const float* __restrict__ in,
                              unsigned short* __restrict__ out,
                              int K, int N) {
    long long i = (long long)blockIdx.x * blockDim.x + threadIdx.x;
    if (i < (long long)K * N) {
        int n = (int)(i / K);
        int k = (int)(i % K);
        out[i] = f2bf(in[(long long)k * N + n]);
    }
}

// ---------------------------------------------------------------------------
// C[M][N] = A[M][Kd](bf16) * Wt[N][Kd](bf16)^T + bias
// 128 threads (4 waves). Block tile 128x64, K-step 32, double-buffered LDS
// filled with async global->LDS b128 copies. Each wave computes a 32x64
// patch: 2 A-frags x 4 B-frags -> 8 WMMAs per K-step.
// ---------------------------------------------------------------------------
#define BM 128
#define BN 64
#define BK 32

__global__ __launch_bounds__(128)
void gemm_bf16(const unsigned short* __restrict__ A,
               const unsigned short* __restrict__ Bt,
               const float* __restrict__ bias,
               void* __restrict__ Cout,
               int M, int N, int Kd, int out_f32) {
    __shared__ uint4 Abuf[2][BM * BK / 8]; // 128 rows x 32 bf16 = 8 KB each
    __shared__ uint4 Bbuf[2][BN * BK / 8]; // 64 rows x 32 bf16 = 4 KB each

    const int tid  = threadIdx.x;
    const int lane = tid & 31;
    const int wave = tid >> 5;
    const int l16  = lane & 15;
    const int g    = lane >> 4;
    const int m0   = blockIdx.y * BM;
    const int n0   = blockIdx.x * BN;

    // per-thread async-fill addressing:
    //   A: thread t copies its whole 64B row slice (4 x b128)
    //   B: thread t copies half of row t/2 (2 x b128)
    const int brow = tid >> 1;
    const int bhalf = tid & 1;
    const char* Abase = (const char*)A + (size_t)(m0 + tid) * Kd * 2;
    const char* Bbase =
        (const char*)Bt + ((size_t)(n0 + brow) * Kd + bhalf * 16) * 2;

    auto issue_tile = [&](int buf, int k0) {
        unsigned al = lds_off(&Abuf[buf][tid * 4]);
        unsigned long long ag = (unsigned long long)(Abase + (size_t)k0 * 2);
#pragma unroll
        for (int j = 0; j < 4; ++j)
            async_copy_b128(al + j * 16, ag + j * 16);
        unsigned bl = lds_off(&Bbuf[buf][brow * 4 + bhalf * 2]);
        unsigned long long bg = (unsigned long long)(Bbase + (size_t)k0 * 2);
#pragma unroll
        for (int j = 0; j < 2; ++j)
            async_copy_b128(bl + j * 16, bg + j * 16);
    };

    v8f acc[2][4] = {};

    issue_tile(0, 0);
    wait_async0();
    __syncthreads();

    int buf = 0;
    for (int k0 = 0; k0 < Kd; k0 += BK) {
        if (k0 + BK < Kd) issue_tile(buf ^ 1, k0 + BK);

        const unsigned int* As = (const unsigned int*)&Abuf[buf][0]; // [128][16]
        const unsigned int* Bs = (const unsigned int*)&Bbuf[buf][0]; // [64][16]

        // A fragments: 16x32 bf16; lane half g: VGPR i holds K pair
        // k = (i<4 ? 0 : 16) + g*8 + 2*(i&3)
        AFrag a[2];
#pragma unroll
        for (int mt = 0; mt < 2; ++mt)
#pragma unroll
            for (int i = 0; i < 8; ++i) {
                int k = ((i < 4) ? 0 : 16) + g * 8 + (i & 3) * 2;
                a[mt].w[i] =
                    As[(wave * 32 + mt * 16 + l16) * 16 + (k >> 1)];
            }

#pragma unroll
        for (int nt = 0; nt < 4; ++nt) {
            // B fragment: 32x16; lane n = l16, VGPR v holds pair k = (v<4 ? 0 : 16) + g*8 + 2*(v&3)
            AFrag bfr;
#pragma unroll
            for (int v = 0; v < 8; ++v)
                bfr.w[v] = Bs[(nt * 16 + l16) * 16 + (((v < 4) ? 0 : 8) + g * 4 + (v & 3))];
            acc[0][nt] = __builtin_amdgcn_wmma_f32_16x16x32_bf16(
                false, a[0].v, false, bfr.v, (short)0, acc[0][nt], false, false);
            acc[1][nt] = __builtin_amdgcn_wmma_f32_16x16x32_bf16(
                false, a[1].v, false, bfr.v, (short)0, acc[1][nt], false, false);
        }

        wait_async0();
        __syncthreads();
        buf ^= 1;
    }

    // epilogue: C/D layout -> element (M = v + 8*g, N = l16) per VGPR v
#pragma unroll
    for (int nt = 0; nt < 4; ++nt) {
        int nglob = n0 + nt * 16 + l16;
        float bv = bias[nglob];
#pragma unroll
        for (int mt = 0; mt < 2; ++mt)
#pragma unroll
            for (int v = 0; v < 8; ++v) {
                int mglob = m0 + wave * 32 + mt * 16 + v + 8 * g;
                float c = acc[mt][nt][v] + bv;
                if (out_f32)
                    ((float*)Cout)[(size_t)mglob * N + nglob] = c;
                else
                    ((unsigned short*)Cout)[(size_t)mglob * N + nglob] =
                        f2bf(c);
            }
    }
}

// ---------------------------------------------------------------------------
// Flash-style GQA attention.
// Workgroup = 128 threads = 4 waves = the 4 q-heads sharing one KV head.
// Grid: (q-tile, kv-head, batch). K/V 32-key chunks are async-copied to LDS
// once per workgroup (shared by all 4 waves) and double-buffered.
// Q: [B*S][HIDDEN] bf16 (col qh*64+d), K/V: [B*S][KV_DIM] bf16.
// H out: [B*S][HIDDEN] bf16.
// ---------------------------------------------------------------------------
__global__ __launch_bounds__(128)
void attn_kernel(const unsigned short* __restrict__ Qb,
                 const unsigned short* __restrict__ Kb,
                 const unsigned short* __restrict__ Vb,
                 unsigned short* __restrict__ Hb) {
    __shared__ unsigned int Klds[2][32 * 32]; // 32 keys x 64 bf16, 4 KB each
    __shared__ unsigned int Vlds[2][32 * 32]; // 32 keys x 64 bf16, 4 KB each
    __shared__ unsigned int Qlds[4][16 * 32]; // per-wave 16 x 64 bf16
    __shared__ unsigned int Plds[4][16 * 16]; // per-wave 16 x 32 bf16

    const int tid  = threadIdx.x;
    const int lane = tid & 31;
    const int wave = tid >> 5;
    const int l16  = lane & 15;
    const int g    = lane >> 4;
    const int qt   = blockIdx.x;            // query tile 0..127
    const int kvh  = blockIdx.y;            // kv head 0..7
    const int b    = blockIdx.z;            // batch
    const int qh   = kvh * 4 + wave;        // this wave's q head
    const size_t qrow0 = (size_t)(b * SEQ + qt * 16);
    const size_t kvrow0 = (size_t)(b * SEQ); // + key

    // async fill of one 32-key K/V chunk: 2 b128 per thread for each of K, V
    auto issue_kv = [&](int bufi, int c0) {
#pragma unroll
        for (int j = 0; j < 2; ++j) {
            int idx = tid * 2 + j;          // 0..255
            int r = idx >> 3;               // key row 0..31
            int c = idx & 7;                // 16B chunk in row (64 bf16=128B)
            size_t gofs =
                ((kvrow0 + c0 + r) * KV_DIM + (size_t)kvh * 64) * 2 + c * 16;
            async_copy_b128(lds_off(&Klds[bufi][0]) + idx * 16,
                            (unsigned long long)((const char*)Kb + gofs));
            async_copy_b128(lds_off(&Vlds[bufi][0]) + idx * 16,
                            (unsigned long long)((const char*)Vb + gofs));
        }
    };

    // stage this wave's Q tile (16 x 64 bf16 = 512 uints); wave-private
    const unsigned int* Qg = (const unsigned int*)Qb;
    for (int j = 0; j < 16; ++j) {
        int idx = lane + j * 32;
        int r = idx >> 5;
        int c = idx & 31;
        Qlds[wave][idx] = Qg[(qrow0 + r) * (HIDDEN / 2) + (size_t)qh * 32 + c];
    }

    issue_kv(0, 0);

    // two A fragments of Q (d 0..31 and 32..63); in-wave LDS ordering
    AFrag aq[2];
#pragma unroll
    for (int ds = 0; ds < 2; ++ds)
#pragma unroll
        for (int i = 0; i < 8; ++i) {
            int d = ds * 32 + ((i < 4) ? 0 : 16) + g * 8 + (i & 3) * 2;
            aq[ds].w[i] = Qlds[wave][l16 * 32 + (d >> 1)];
        }

    float mrow[8], lrow[8];
#pragma unroll
    for (int v = 0; v < 8; ++v) { mrow[v] = -1e30f; lrow[v] = 0.0f; }
    v8f acc[4] = {};

    const float scale = 0.125f; // 1/sqrt(64)

    wait_async0();
    __syncthreads();

    int buf = 0;
    for (int c0 = 0; c0 < SEQ; c0 += 32) {
        if (c0 + 32 < SEQ) issue_kv(buf ^ 1, c0 + 32);

        const unsigned int* Ks = &Klds[buf][0];       // [32 keys][32 uints]
        const unsigned short* Vs =
            (const unsigned short*)&Vlds[buf][0];     // [32 keys][64 bf16]

        // ---- scores: 16 q-rows x 32 keys = two 16x16 tiles, K-dim 64 ----
        v8f s[2] = {};
#pragma unroll
        for (int nt = 0; nt < 2; ++nt) {
#pragma unroll
            for (int ds = 0; ds < 2; ++ds) {
                AFrag bk; // K^T fragment: n = key, k = d
#pragma unroll
                for (int v = 0; v < 8; ++v) {
                    int key = nt * 16 + l16;
                    int dc = ds * 16 + ((v < 4) ? 0 : 8) + g * 4 + (v & 3); // uint col = d/2
                    bk.w[v] = Ks[key * 32 + dc];
                }
                s[nt] = __builtin_amdgcn_wmma_f32_16x16x32_bf16(
                    false, aq[ds].v, false, bk.v, (short)0, s[nt], false, false);
            }
        }

        // ---- online softmax (row lives in one VGPR across a 16-lane half) --
        float p0[8], p1[8];
#pragma unroll
        for (int v = 0; v < 8; ++v) {
            float x0 = s[0][v] * scale;
            float x1 = s[1][v] * scale;
            float mx = fmaxf(x0, x1);
#pragma unroll
            for (int off = 8; off >= 1; off >>= 1)
                mx = fmaxf(mx, __shfl_xor(mx, off, 16));
            float mnew = fmaxf(mrow[v], mx);
            float alpha = __expf(mrow[v] - mnew);
            float e0 = __expf(x0 - mnew);
            float e1 = __expf(x1 - mnew);
            float sum = e0 + e1;
#pragma unroll
            for (int off = 8; off >= 1; off >>= 1)
                sum += __shfl_xor(sum, off, 16);
            lrow[v] = lrow[v] * alpha + sum;
            mrow[v] = mnew;
            p0[v] = e0; p1[v] = e1;
#pragma unroll
            for (int nt = 0; nt < 4; ++nt) acc[nt][v] *= alpha;
        }

        // ---- P (C-layout f32) -> wave-private LDS bf16 -> A-fragment ------
        {
            unsigned short* P16 = (unsigned short*)&Plds[wave][0];
#pragma unroll
            for (int v = 0; v < 8; ++v) {
                int row = v + 8 * g;
                P16[row * 32 + l16]      = f2bf(p0[v]);
                P16[row * 32 + 16 + l16] = f2bf(p1[v]);
            }
        }
        AFrag pa; // in-wave DS ordering makes the writes visible
#pragma unroll
        for (int i = 0; i < 8; ++i) {
            int k = ((i < 4) ? 0 : 16) + g * 8 + (i & 3) * 2;
            pa.w[i] = Plds[wave][l16 * 16 + (k >> 1)];
        }

        // ---- acc += P(16x32) x V(32x64) ----
#pragma unroll
        for (int nt = 0; nt < 4; ++nt) {
            AFrag bv; // V fragment: n = d col, k = key (row stride 64 bf16)
#pragma unroll
            for (int v = 0; v < 8; ++v) {
                int key = ((v < 4) ? 0 : 16) + g * 8 + (v & 3) * 2;
                int d = nt * 16 + l16;
                bv.u[2 * v]     = Vs[key * 64 + d];
                bv.u[2 * v + 1] = Vs[(key + 1) * 64 + d];
            }
            acc[nt] = __builtin_amdgcn_wmma_f32_16x16x32_bf16(
                false, pa.v, false, bv.v, (short)0, acc[nt], false, false);
        }

        wait_async0();
        __syncthreads();
        buf ^= 1;
    }

    // normalize and write H[b, s, qh*64 + d] (bf16)
#pragma unroll
    for (int nt = 0; nt < 4; ++nt) {
#pragma unroll
        for (int v = 0; v < 8; ++v) {
            int srow = qt * 16 + v + 8 * g;
            float o = acc[nt][v] / lrow[v];
            Hb[(size_t)(b * SEQ + srow) * HIDDEN + qh * 64 + nt * 16 + l16] =
                f2bf(o);
        }
    }
}

// ---------------------------------------------------------------------------
extern "C" void kernel_launch(void* const* d_in, const int* in_sizes, int n_in,
                              void* d_out, int out_size, void* d_ws,
                              size_t ws_size, hipStream_t stream) {
    (void)in_sizes; (void)n_in; (void)out_size; (void)ws_size;
    const float* X  = (const float*)d_in[0];
    const float* wq = (const float*)d_in[1];
    const float* bq = (const float*)d_in[2];
    const float* wk = (const float*)d_in[3];
    const float* bk = (const float*)d_in[4];
    const float* wv = (const float*)d_in[5];
    const float* bv = (const float*)d_in[6];
    const float* wo = (const float*)d_in[7];
    const float* bo = (const float*)d_in[8];
    float* out = (float*)d_out;

    char* ws = (char*)d_ws;
    size_t off = 0;
    auto alloc = [&](size_t bytes) -> char* {
        char* p = ws + off;
        off += (bytes + 255) & ~(size_t)255;
        return p;
    };
    unsigned short* Xh  = (unsigned short*)alloc((size_t)MTOT * HIDDEN * 2);
    unsigned short* wqt = (unsigned short*)alloc((size_t)HIDDEN * HIDDEN * 2);
    unsigned short* wkt = (unsigned short*)alloc((size_t)KV_DIM * HIDDEN * 2);
    unsigned short* wvt = (unsigned short*)alloc((size_t)KV_DIM * HIDDEN * 2);
    unsigned short* wot = (unsigned short*)alloc((size_t)HIDDEN * HIDDEN * 2);
    unsigned short* Qb  = (unsigned short*)alloc((size_t)MTOT * HIDDEN * 2);
    unsigned short* Kb  = (unsigned short*)alloc((size_t)MTOT * KV_DIM * 2);
    unsigned short* Vb  = (unsigned short*)alloc((size_t)MTOT * KV_DIM * 2);
    unsigned short* Hb  = (unsigned short*)alloc((size_t)MTOT * HIDDEN * 2);

    const int nX = MTOT * HIDDEN;
    cvt_f32_bf16<<<(nX + 255) / 256, 256, 0, stream>>>(X, Xh, nX);
    transpose_cvt<<<(HIDDEN * HIDDEN + 255) / 256, 256, 0, stream>>>(
        wq, wqt, HIDDEN, HIDDEN);
    transpose_cvt<<<(HIDDEN * KV_DIM + 255) / 256, 256, 0, stream>>>(
        wk, wkt, HIDDEN, KV_DIM);
    transpose_cvt<<<(HIDDEN * KV_DIM + 255) / 256, 256, 0, stream>>>(
        wv, wvt, HIDDEN, KV_DIM);
    transpose_cvt<<<(HIDDEN * HIDDEN + 255) / 256, 256, 0, stream>>>(
        wo, wot, HIDDEN, HIDDEN);

    dim3 blk(128);
    dim3 gq(HIDDEN / BN, MTOT / BM);   // (32, 32)
    dim3 gkv(KV_DIM / BN, MTOT / BM);  // (8, 32)
    gemm_bf16<<<gq, blk, 0, stream>>>(Xh, wqt, bq, Qb, MTOT, HIDDEN, HIDDEN, 0);
    gemm_bf16<<<gkv, blk, 0, stream>>>(Xh, wkt, bk, Kb, MTOT, KV_DIM, HIDDEN, 0);
    gemm_bf16<<<gkv, blk, 0, stream>>>(Xh, wvt, bv, Vb, MTOT, KV_DIM, HIDDEN, 0);

    dim3 ga(SEQ / 16, NUM_KV_HEAD, BATCH);
    attn_kernel<<<ga, blk, 0, stream>>>(Qb, Kb, Vb, Hb);

    gemm_bf16<<<gq, blk, 0, stream>>>(Hb, wot, bo, (void*)out, MTOT, HIDDEN,
                                      HIDDEN, 1);
}
